// LightNetAttention_15015205667340
// MI455X (gfx1250) — matrix operand order repaired and checked by reading, not compiled, not run on hardware
//
#include <hip/hip_runtime.h>
#include <hip/hip_bf16.h>
#include <math.h>

// ---------------------------------------------------------------------------
// LightNet attention (GLA-style) for MI455X / gfx1250, wave32.
// Big GEMMs: v_wmma_f32_16x16x32_bf16 (bf16 in, f32 accum), double-buffered LDS.
// GLA recurrence: v_wmma_f32_16x16x4_f32 (full fp32, no conversion loss).
// ---------------------------------------------------------------------------

typedef __attribute__((ext_vector_type(16))) __bf16 v16bf;
typedef __attribute__((ext_vector_type(8)))  __bf16 v8bf;
typedef __attribute__((ext_vector_type(8)))  float  v8f;
typedef __attribute__((ext_vector_type(2)))  float  v2f;

#define BB 4
#define TT 2048
#define DDIM 1024
#define MM (BB * TT)          // 8192 rows
#define HH 8
#define DKK 128
#define CCHUNK 64
#define NCHUNK (TT / CCHUNK)  // 32

static __device__ __forceinline__ float fast_sigmoid(float x) {
  // v_rcp_f32 instead of IEEE-division lowering (saves ~12 VALU/element)
  return __builtin_amdgcn_rcpf(1.0f + __expf(-x));
}

static __device__ __forceinline__ v8f v8f_zero() {
  v8f r;
#pragma unroll
  for (int i = 0; i < 8; ++i) r[i] = 0.0f;
  return r;
}

// ---------------------------------------------------------------------------
// fp32 -> bf16 conversion (grid-stride)
// ---------------------------------------------------------------------------
__global__ void cvt_bf16_kernel(const float* __restrict__ in,
                                __bf16* __restrict__ out, size_t n) {
  size_t i = (size_t)blockIdx.x * blockDim.x + threadIdx.x;
  size_t stride = (size_t)gridDim.x * blockDim.x;
  for (; i < n; i += stride) out[i] = (__bf16)in[i];
}

// ---------------------------------------------------------------------------
// Generic GEMM: C[M,N] = act( A[M,K] @ W[N,K]^T ), A/W bf16, C f32.
// Block: 256 thr (8 waves), 128x128 C tile, K-step 32, double-buffered LDS.
// Wave grid 2x4; each wave 64x32 => 4x2 subtiles of 16x16.
// ACT==1: swish epilogue (compile-time, no branch in unrolled epilogue).
// ---------------------------------------------------------------------------
#define GST 40   // LDS row stride in bf16 (32 data + 8 pad, keeps 16B align)

template <int ACT>
__global__ __launch_bounds__(256) void gemm_bf16_nt(
    const __bf16* __restrict__ A, const __bf16* __restrict__ W,
    float* __restrict__ C, int M, int N, int K) {
  __shared__ __bf16 As[2][128 * GST];
  __shared__ __bf16 Bs[2][128 * GST];

  const int tid  = threadIdx.x;
  const int lane = tid & 31;
  const int wave = tid >> 5;
  const int gm0  = blockIdx.y * 128;
  const int gn0  = blockIdx.x * 128;
  const int wm   = (wave >> 2) * 64;
  const int wn   = (wave & 3) * 32;

  v8f acc[4][2];
#pragma unroll
  for (int i = 0; i < 4; ++i)
#pragma unroll
    for (int j = 0; j < 2; ++j) acc[i][j] = v8f_zero();

  const int r  = tid >> 1;            // 0..127 (loads)
  const int cs = (tid & 1) << 4;      // 0 or 16
  const __bf16* gaBase = A + (size_t)(gm0 + r) * K + cs;
  const __bf16* gbBase = W + (size_t)(gn0 + r) * K + cs;

  // prologue: stage chunk 0 into buffer 0
  {
    const uint4* ga = (const uint4*)gaBase;
    const uint4* gb = (const uint4*)gbBase;
    uint4 a0 = ga[0], a1 = ga[1], b0 = gb[0], b1 = gb[1];
    uint4* la = (uint4*)(&As[0][r * GST + cs]); la[0] = a0; la[1] = a1;
    uint4* lb = (uint4*)(&Bs[0][r * GST + cs]); lb[0] = b0; lb[1] = b1;
  }

  const int nch = K >> 5;
  for (int ci = 0; ci < nch; ++ci) {
    __syncthreads();  // LDS[cur] ready (and previous stores to it complete)
    const int cur = ci & 1;
    const int nxt = cur ^ 1;

    // Issue next chunk's global loads early (latency hidden under WMMAs)
    uint4 na0, na1, nb0, nb1;
    const bool more = (ci + 1) < nch;
    if (more) {
      const uint4* ga = (const uint4*)(gaBase + ((ci + 1) << 5));
      const uint4* gb = (const uint4*)(gbBase + ((ci + 1) << 5));
      na0 = ga[0]; na1 = ga[1]; nb0 = gb[0]; nb1 = gb[1];
    }

    // B fragments (32x16 bf16): N=lane&15; K = kb2+0..7 and 16+kb2+0..7 (same as A)
    v16bf bfrag[2];
#pragma unroll
    for (int sn = 0; sn < 2; ++sn) {
      const __bf16* bp =
          &Bs[cur][(wn + sn * 16 + (lane & 15)) * GST + ((lane >> 4) << 3)];
      v8bf lo = *(const v8bf*)bp;
      v8bf hi = *(const v8bf*)(bp + 16);
#pragma unroll
      for (int j = 0; j < 8; ++j) { bfrag[sn][j] = lo[j]; bfrag[sn][8 + j] = hi[j]; }
    }

#pragma unroll
    for (int sm = 0; sm < 4; ++sm) {
      // A fragment (16x32 bf16): row = lane&15; K = kb2+0..7 and 16+kb2+0..7
      const __bf16* ap =
          &As[cur][(wm + sm * 16 + (lane & 15)) * GST + ((lane >> 4) << 3)];
      v8bf lo = *(const v8bf*)ap;
      v8bf hi = *(const v8bf*)(ap + 16);
      v16bf afrag;
#pragma unroll
      for (int j = 0; j < 8; ++j) { afrag[j] = lo[j]; afrag[8 + j] = hi[j]; }
#pragma unroll
      for (int sn = 0; sn < 2; ++sn)
        acc[sm][sn] = __builtin_amdgcn_wmma_f32_16x16x32_bf16(
            false, afrag, false, bfrag[sn], (short)0, acc[sm][sn], false, false);
    }

    // Stage next chunk into the other buffer (no conflict with cur readers)
    if (more) {
      uint4* la = (uint4*)(&As[nxt][r * GST + cs]); la[0] = na0; la[1] = na1;
      uint4* lb = (uint4*)(&Bs[nxt][r * GST + cs]); lb[0] = nb0; lb[1] = nb1;
    }
  }

  // Epilogue (C/D layout: lane<16 -> M=vgpr, N=lane; lane>=16 -> M=8+vgpr)
  const int nn = lane & 15;
  const int mb = (lane >> 4) << 3;
#pragma unroll
  for (int sm = 0; sm < 4; ++sm)
#pragma unroll
    for (int sn = 0; sn < 2; ++sn) {
      const int m0 = gm0 + wm + sm * 16 + mb;
      const int n0 = gn0 + wn + sn * 16 + nn;
#pragma unroll
      for (int rr = 0; rr < 8; ++rr) {
        float vo = acc[sm][sn][rr];
        if (ACT == 1) vo = vo * fast_sigmoid(vo);
        C[(size_t)(m0 + rr) * N + n0] = vo;
      }
    }
}

// ---------------------------------------------------------------------------
// cumlogsumexp over T per (b, channel). 4096 independent channels.
// ---------------------------------------------------------------------------
__global__ __launch_bounds__(256) void cumlse_kernel(
    const float* __restrict__ kraw, float* __restrict__ z) {
  const int g = blockIdx.x * 256 + threadIdx.x;  // 0..4095
  const int b = g >> 10;
  const int c = g & 1023;
  const float* p = kraw + (size_t)b * TT * DDIM + c;
  float* zo = z + (size_t)b * TT * DDIM + c;
  float m = -INFINITY, s = 0.0f;
  for (int t = 0; t < TT; ++t) {
    const float x = p[(size_t)t * DDIM];
    if (x > m) { s = s * __expf(m - x) + 1.0f; m = x; }
    else       { s += __expf(x - m); }
    zo[(size_t)t * DDIM] = m + __logf(s);
  }
}

// ---------------------------------------------------------------------------
// Chunked GLA recurrence. Block = (vb, h, b): one (b,h) head, half of DV.
// 256 threads / 8 waves, sequential over 32 chunks, state S in LDS (f32).
// All matmuls via V_WMMA_F32_16X16X4_F32.
// Telescoped factors:
//   q2 = q * exp(zprev - z);  k2 = exp(kraw - zprev)
//   kT = exp(kraw - zlast);   decay = exp(zprev - zlast)
// ---------------------------------------------------------------------------
#define SQ 129   // q2/k2 row stride (128 + 1 pad)
#define SC 65    // 64-wide tiles row stride (64 + 1 pad)
#define GLA_LDS_FLOATS (2 * 64 * SQ + 128 * SC + 64 * SC + 128 * SC + 64 * SC + 128)

__global__ __launch_bounds__(256) void gla_kernel(
    const float* __restrict__ q, const float* __restrict__ kraw,
    const float* __restrict__ v, const float* __restrict__ z,
    float* __restrict__ o) {
  extern __shared__ float smem[];
  float* sQ2  = smem;                // 64 x SQ
  float* sK2  = sQ2 + 64 * SQ;       // 64 x SQ
  float* sKT  = sK2 + 64 * SQ;       // 128 x SC  (transposed: [d][c])
  float* sV   = sKT + 128 * SC;      // 64 x SC
  float* sS   = sV + 64 * SC;        // 128 x SC  (state, fp32 master)
  float* sA   = sS + 128 * SC;       // 64 x SC   (masked attn scores)
  float* sDec = sA + 64 * SC;        // 128

  const int tid  = threadIdx.x;
  const int lane = tid & 31;
  const int wave = tid >> 5;
  const int vb   = blockIdx.x;       // 0..1 (DV half)
  const int h    = blockIdx.y;       // 0..7
  const int b    = blockIdx.z;       // 0..3
  const int ch   = h * DKK;
  const int vch  = ch + vb * 64;
  const size_t base = (size_t)b * TT * DDIM;

  const int nn = lane & 15;                 // N (and A-row) index
  const int mb = (lane >> 4) << 3;          // C/D M offset: 0 or 8
  const int kp = (lane >> 4) << 1;          // f32 A/B K parity: 0 or 2

  for (int i = tid; i < 128 * SC; i += 256) sS[i] = 0.0f;

  for (int n = 0; n < NCHUNK; ++n) {
    const int t0 = n * CCHUNK;
    const int tp = (n == 0) ? 0 : (t0 - 1);
    __syncthreads();  // protect tile reuse across chunks

    // ---- Phase L: stage tiles -------------------------------------------
    for (int idx = tid; idx < 64 * 128; idx += 256) {
      const int c = idx >> 7, d = idx & 127;
      const size_t col = base + ch + d;
      const float zc = z[col + (size_t)(t0 + c) * DDIM];
      const float zp = z[col + (size_t)tp * DDIM];
      const float zl = z[col + (size_t)(t0 + 63) * DDIM];
      const float qv = q[col + (size_t)(t0 + c) * DDIM];
      const float kr = kraw[col + (size_t)(t0 + c) * DDIM];
      sQ2[c * SQ + d] = qv * __expf(zp - zc);
      sK2[c * SQ + d] = __expf(kr - zp);
      sKT[d * SC + c] = __expf(kr - zl);
    }
    for (int idx = tid; idx < 64 * 64; idx += 256) {
      const int c = idx >> 6, e = idx & 63;
      sV[c * SC + e] = v[base + (size_t)(t0 + c) * DDIM + vch + e];
    }
    if (tid < 128)
      sDec[tid] = __expf(z[base + (size_t)tp * DDIM + ch + tid] -
                         z[base + (size_t)(t0 + 63) * DDIM + ch + tid]);
    __syncthreads();

    // ---- Phase A: Attn = tril(q2 @ k2^T), 2 subtiles/wave ----------------
#pragma unroll
    for (int s = 0; s < 2; ++s) {
      const int st = wave * 2 + s;
      const int ti = st >> 2, tj = st & 3;
      v8f acc = v8f_zero();
#pragma unroll 4
      for (int kb = 0; kb < 128; kb += 4) {
        v2f a, bb;
        { const float* p = &sQ2[(ti * 16 + nn) * SQ + kb + kp]; a[0] = p[0]; a[1] = p[1]; }
        { const float* p = &sK2[(tj * 16 + nn) * SQ + kb + kp]; bb[0] = p[0]; bb[1] = p[1]; }
        acc = __builtin_amdgcn_wmma_f32_16x16x4_f32(false, a, false, bb,
                                                    (short)0, acc, false, false);
      }
#pragma unroll
      for (int rr = 0; rr < 8; ++rr) {
        const int gi = ti * 16 + mb + rr, gj = tj * 16 + nn;
        sA[gi * SC + gj] = (gj <= gi) ? acc[rr] : 0.0f;
      }
    }

    // ---- Phase B: o_inter = q2 @ S(old), 2 subtiles/wave -----------------
    v8f oacc[2];
#pragma unroll
    for (int s = 0; s < 2; ++s) {
      const int st = wave * 2 + s;
      const int ti = st >> 2, te = st & 3;
      v8f acc = v8f_zero();
#pragma unroll 4
      for (int kb = 0; kb < 128; kb += 4) {
        v2f a, bb;
        { const float* p = &sQ2[(ti * 16 + nn) * SQ + kb + kp]; a[0] = p[0]; a[1] = p[1]; }
        bb[0] = sS[(kb + kp) * SC + te * 16 + nn];
        bb[1] = sS[(kb + kp + 1) * SC + te * 16 + nn];
        acc = __builtin_amdgcn_wmma_f32_16x16x4_f32(false, a, false, bb,
                                                    (short)0, acc, false, false);
      }
      oacc[s] = acc;
    }
    __syncthreads();  // Attn visible; S reads done before decay

    // ---- Phase C: o = o_inter + Attn @ v ; decay S -----------------------
#pragma unroll
    for (int s = 0; s < 2; ++s) {
      const int st = wave * 2 + s;
      const int ti = st >> 2, te = st & 3;
      v8f acc = oacc[s];
#pragma unroll 4
      for (int kb = 0; kb < 64; kb += 4) {
        v2f a, bb;
        { const float* p = &sA[(ti * 16 + nn) * SC + kb + kp]; a[0] = p[0]; a[1] = p[1]; }
        bb[0] = sV[(kb + kp) * SC + te * 16 + nn];
        bb[1] = sV[(kb + kp + 1) * SC + te * 16 + nn];
        acc = __builtin_amdgcn_wmma_f32_16x16x4_f32(false, a, false, bb,
                                                    (short)0, acc, false, false);
      }
#pragma unroll
      for (int rr = 0; rr < 8; ++rr)
        o[base + (size_t)(t0 + ti * 16 + mb + rr) * DDIM + vch + te * 16 + nn] = acc[rr];
    }
    for (int idx = tid; idx < 128 * 64; idx += 256) {
      const int d = idx >> 6, e = idx & 63;
      sS[d * SC + e] *= sDec[d];
    }
    __syncthreads();

    // ---- Phase D: S += kT^T @ v, 4 subtiles/wave -------------------------
#pragma unroll
    for (int s = 0; s < 4; ++s) {
      const int st = wave * 4 + s;
      const int td = st >> 2, te = st & 3;
      v8f acc;
#pragma unroll
      for (int rr = 0; rr < 8; ++rr)
        acc[rr] = sS[(td * 16 + mb + rr) * SC + te * 16 + nn];
#pragma unroll 4
      for (int kb = 0; kb < 64; kb += 4) {
        v2f a, bb;
        { const float* p = &sKT[(td * 16 + nn) * SC + kb + kp]; a[0] = p[0]; a[1] = p[1]; }
        bb[0] = sV[(kb + kp) * SC + te * 16 + nn];
        bb[1] = sV[(kb + kp + 1) * SC + te * 16 + nn];
        acc = __builtin_amdgcn_wmma_f32_16x16x4_f32(false, a, false, bb,
                                                    (short)0, acc, false, false);
      }
#pragma unroll
      for (int rr = 0; rr < 8; ++rr)
        sS[(td * 16 + mb + rr) * SC + te * 16 + nn] = acc[rr];
    }
  }
}

// ---------------------------------------------------------------------------
// y[r,:] = rmsnorm(o[r,:]) * g_norm_w * swish(gate[r,:])  -> bf16
// One block per row.
// ---------------------------------------------------------------------------
__global__ __launch_bounds__(256) void prep_kernel(
    const float* __restrict__ o, const float* __restrict__ gate,
    const float* __restrict__ gw, __bf16* __restrict__ y) {
  const int r = blockIdx.x;
  const int tid = threadIdx.x;
  const float* orow = o + (size_t)r * DDIM;
  const float* grow = gate + (size_t)r * DDIM;
  __bf16* yrow = y + (size_t)r * DDIM;

  float s = 0.0f;
  for (int d = tid; d < DDIM; d += 256) { const float t = orow[d]; s += t * t; }
  __shared__ float red[256];
  red[tid] = s;
  __syncthreads();
  for (int off = 128; off > 0; off >>= 1) {
    if (tid < off) red[tid] += red[tid + off];
    __syncthreads();
  }
  const float rinv = rsqrtf(red[0] * (1.0f / DDIM) + 1e-5f);
  for (int d = tid; d < DDIM; d += 256) {
    const float g = grow[d];
    yrow[d] = (__bf16)(orow[d] * rinv * gw[d] * g * fast_sigmoid(g));
  }
}

// ---------------------------------------------------------------------------
// Host-side orchestration
// ---------------------------------------------------------------------------
extern "C" void kernel_launch(void* const* d_in, const int* in_sizes, int n_in,
                              void* d_out, int out_size, void* d_ws, size_t ws_size,
                              hipStream_t stream) {
  (void)in_sizes; (void)n_in; (void)out_size; (void)ws_size;
  const float* x    = (const float*)d_in[0];
  const float* Wq   = (const float*)d_in[1];
  const float* Wk   = (const float*)d_in[2];
  const float* Wv   = (const float*)d_in[3];
  const float* Wg1  = (const float*)d_in[4];
  const float* Wg2  = (const float*)d_in[5];
  const float* gnw  = (const float*)d_in[6];
  const float* Wo   = (const float*)d_in[7];

  uint8_t* w = (uint8_t*)d_ws;
  size_t off = 0;
  auto alloc = [&](size_t bytes) -> void* {
    void* p = w + off;
    off += (bytes + 255) & ~(size_t)255;
    return p;
  };
  const size_t MD = (size_t)MM * DDIM;
  __bf16* xb   = (__bf16*)alloc(MD * 2);
  __bf16* wqb  = (__bf16*)alloc((size_t)DDIM * DDIM * 2);
  __bf16* wkb  = (__bf16*)alloc((size_t)DDIM * DDIM * 2);
  __bf16* wvb  = (__bf16*)alloc((size_t)DDIM * DDIM * 2);
  __bf16* wob  = (__bf16*)alloc((size_t)DDIM * DDIM * 2);
  __bf16* wg1b = (__bf16*)alloc((size_t)128 * DDIM * 2);
  __bf16* wg2b = (__bf16*)alloc((size_t)DDIM * 128 * 2);
  float*  qf   = (float*)alloc(MD * 4);
  float*  kf   = (float*)alloc(MD * 4);
  float*  vf   = (float*)alloc(MD * 4);
  float*  zf   = (float*)alloc(MD * 4);
  float*  g1f  = (float*)alloc((size_t)MM * 128 * 4);
  float*  gatf = (float*)alloc(MD * 4);
  float*  of   = (float*)alloc(MD * 4);
  __bf16* g1b  = (__bf16*)alloc((size_t)MM * 128 * 2);
  __bf16* yb   = (__bf16*)alloc(MD * 2);

  // bf16 conversions
  cvt_bf16_kernel<<<2048, 256, 0, stream>>>(x, xb, MD);
  cvt_bf16_kernel<<<512, 256, 0, stream>>>(Wq, wqb, (size_t)DDIM * DDIM);
  cvt_bf16_kernel<<<512, 256, 0, stream>>>(Wk, wkb, (size_t)DDIM * DDIM);
  cvt_bf16_kernel<<<512, 256, 0, stream>>>(Wv, wvb, (size_t)DDIM * DDIM);
  cvt_bf16_kernel<<<512, 256, 0, stream>>>(Wo, wob, (size_t)DDIM * DDIM);
  cvt_bf16_kernel<<<128, 256, 0, stream>>>(Wg1, wg1b, (size_t)128 * DDIM);
  cvt_bf16_kernel<<<128, 256, 0, stream>>>(Wg2, wg2b, (size_t)DDIM * 128);

  // Projections
  dim3 gFull(DDIM / 128, MM / 128);   // (8, 64)
  gemm_bf16_nt<1><<<gFull, 256, 0, stream>>>(xb, wqb, qf, MM, DDIM, DDIM);  // swish
  gemm_bf16_nt<0><<<gFull, 256, 0, stream>>>(xb, wkb, kf, MM, DDIM, DDIM);
  gemm_bf16_nt<0><<<gFull, 256, 0, stream>>>(xb, wvb, vf, MM, DDIM, DDIM);
  dim3 gG1(1, MM / 128);
  gemm_bf16_nt<0><<<gG1, 256, 0, stream>>>(xb, wg1b, g1f, MM, 128, DDIM);
  cvt_bf16_kernel<<<512, 256, 0, stream>>>(g1f, g1b, (size_t)MM * 128);
  gemm_bf16_nt<0><<<gFull, 256, 0, stream>>>(g1b, wg2b, gatf, MM, DDIM, 128);

  // cumulative logsumexp over T
  cumlse_kernel<<<16, 256, 0, stream>>>(kf, zf);

  // chunked GLA (one block per (b, h, DV-half))
  const size_t glaShm = (size_t)GLA_LDS_FLOATS * sizeof(float);  // ~166 KB (<320KB/WGP)
  (void)hipFuncSetAttribute((const void*)gla_kernel,
                            hipFuncAttributeMaxDynamicSharedMemorySize,
                            (int)glaShm);
  dim3 gGla(2, HH, BB);
  gla_kernel<<<gGla, 256, glaShm, stream>>>(qf, kf, vf, zf, of);

  // rmsnorm * swish(gate) -> bf16
  prep_kernel<<<MM, 256, 0, stream>>>(of, gatf, gnw, yb);

  // output projection into d_out (f32)
  gemm_bf16_nt<0><<<gFull, 256, 0, stream>>>(yb, wob, (float*)d_out, MM, DDIM, DDIM);
}
